// GPT_KV_33148557590900
// MI455X (gfx1250) — matrix operand order repaired and checked by reading, not compiled, not run on hardware
//
#include <hip/hip_runtime.h>
#include <math.h>

// ---------------------------------------------------------------------------
// GPT forward (B=2, T=2048, L=8, D=1024, H=16, DFF=4096, V=50257) for gfx1250.
// bf16 WMMA GEMMs fed by a double-buffered Tensor Data Mover pipeline
// (tensor_load_to_lds with LDS padding, DMA overlapped with WMMA compute),
// flash attention with WMMA, fused epilogues incl. transposed V layout.
// ---------------------------------------------------------------------------

typedef __bf16 bf16;
typedef __attribute__((ext_vector_type(8)))  bf16  bf16x8;   // 16 bytes
typedef __attribute__((ext_vector_type(16))) bf16  v16bf;    // WMMA A/B frag
typedef __attribute__((ext_vector_type(8)))  float v8f;      // WMMA C/D frag
typedef __attribute__((ext_vector_type(4)))  unsigned int uint4v;  // D# group0
typedef __attribute__((ext_vector_type(8)))  unsigned int uint8v;  // D# group1

#define L_   8
#define D_   1024
#define H_   16
#define HD_  64
#define DFF_ 4096
#define V_   50257
#define B_   2
#define T_   2048
#define M_   (B_*T_)
#define EPS_ 1e-5f

__device__ __forceinline__ float gelu_f(float x) {
    const float c = 0.7978845608028654f; // sqrt(2/pi)
    float t = tanhf(c * (x + 0.044715f * x * x * x));
    return 0.5f * x * (1.0f + t);
}

// ------------------------- fp32 -> bf16 conversion -------------------------
__global__ void cvt_f32_bf16_k(const float* __restrict__ s, bf16* __restrict__ d,
                               long long n) {
    long long i = (long long)blockIdx.x * blockDim.x + threadIdx.x;
    long long stride = (long long)gridDim.x * blockDim.x;
    for (; i < n; i += stride) d[i] = (bf16)s[i];
}

// ------------------------------ embedding ----------------------------------
__global__ void embed_k(const int* __restrict__ x, const float* __restrict__ We,
                        const float* __restrict__ pe, float* __restrict__ h) {
    long long i = (long long)blockIdx.x * blockDim.x + threadIdx.x;
    long long stride = (long long)gridDim.x * blockDim.x;
    const long long n = (long long)M_ * D_;
    for (; i < n; i += stride) {
        int r = (int)(i / D_), d = (int)(i % D_);
        int t = r % T_;
        h[i] = We[(long long)x[r] * D_ + d] + pe[(long long)t * D_ + d];
    }
}

// ------------------- LayerNorm (ddof=1 std, (std+eps)) ---------------------
__global__ __launch_bounds__(256)
void ln_bf16_k(const float* __restrict__ h, const float* __restrict__ sc,
               const float* __restrict__ sh, bf16* __restrict__ out) {
    int row = blockIdx.x;
    const float* hr = h + (long long)row * D_;
    float x[4], s = 0.f, ss = 0.f;
#pragma unroll
    for (int i = 0; i < 4; i++) {
        x[i] = hr[threadIdx.x + i * 256];
        s += x[i]; ss += x[i] * x[i];
    }
    for (int m = 1; m < 32; m <<= 1) { s += __shfl_xor(s, m, 32); ss += __shfl_xor(ss, m, 32); }
    __shared__ float rs[8], rss[8];
    int wid = threadIdx.x >> 5, lane = threadIdx.x & 31;
    if (lane == 0) { rs[wid] = s; rss[wid] = ss; }
    __syncthreads();
    s = 0.f; ss = 0.f;
#pragma unroll
    for (int i = 0; i < 8; i++) { s += rs[i]; ss += rss[i]; }
    float mean = s / D_;
    float var = (ss - D_ * mean * mean) / (float)(D_ - 1);
    float denom = sqrtf(fmaxf(var, 0.f)) + EPS_;
    bf16* orow = out + (long long)row * D_;
#pragma unroll
    for (int i = 0; i < 4; i++) {
        int d = threadIdx.x + i * 256;
        orow[d] = (bf16)(sc[d] * ((x[i] - mean) / denom) + sh[d]);
    }
}

__global__ __launch_bounds__(256)
void ln_final_k(const float* __restrict__ h, const float* __restrict__ sc,
                const float* __restrict__ sh, float* __restrict__ out) {
    int b = blockIdx.x;                              // last token of each batch
    const float* hr = h + ((long long)(b + 1) * T_ - 1) * D_;
    float x[4], s = 0.f, ss = 0.f;
#pragma unroll
    for (int i = 0; i < 4; i++) {
        x[i] = hr[threadIdx.x + i * 256];
        s += x[i]; ss += x[i] * x[i];
    }
    for (int m = 1; m < 32; m <<= 1) { s += __shfl_xor(s, m, 32); ss += __shfl_xor(ss, m, 32); }
    __shared__ float rs[8], rss[8];
    int wid = threadIdx.x >> 5, lane = threadIdx.x & 31;
    if (lane == 0) { rs[wid] = s; rss[wid] = ss; }
    __syncthreads();
    s = 0.f; ss = 0.f;
#pragma unroll
    for (int i = 0; i < 8; i++) { s += rs[i]; ss += rss[i]; }
    float mean = s / D_;
    float var = (ss - D_ * mean * mean) / (float)(D_ - 1);
    float denom = sqrtf(fmaxf(var, 0.f)) + EPS_;
    float* orow = out + (long long)b * D_;
#pragma unroll
    for (int i = 0; i < 4; i++) {
        int d = threadIdx.x + i * 256;
        orow[d] = sc[d] * ((x[i] - mean) / denom) + sh[d];
    }
}

// ------------------- WMMA GEMM (double-buffered TDM) -----------------------
// C[M,N] = A[M,K] * W[N,K]^T  (+ epilogue)
//   mode 0: outb = (bf16)acc                          (q/k projections)
//   mode 1: resid += acc + bias[col]                  (Wo / W2, residual f32 h)
//   mode 2: outb = (bf16)gelu(acc + bias[col])        (W1)
//   mode 3: outb[[B,D,T] transpose] = (bf16)acc       (v projection)
// Workgroup = 256 threads (8 wave32), 128x128 tile; wave tile = 32x64.
// Wave 0 drives the TDM: while tile i is consumed by WMMAs, tile i+1 streams
// into the alternate LDS buffer (descriptor pad_enable reproduces the
// 40-half bank-conflict-free row stride). One barrier per K-step.
__global__ __launch_bounds__(256)
void gemm_wmma_k(const bf16* __restrict__ A, const bf16* __restrict__ W,
                 const float* __restrict__ bias, float* __restrict__ resid,
                 bf16* __restrict__ outb, int Mdim, int Ndim, int Kdim, int mode) {
    __shared__ __align__(16) bf16 As[2][128 * 40];  // 32 used halves/row, stride 40
    __shared__ __align__(16) bf16 Bs[2][128 * 40];
    const int tid = threadIdx.x;
    const int lane = tid & 31, wid = tid >> 5;
    const int wm = wid & 3, wn = wid >> 2;          // 4(M) x 2(N) wave grid
    const int row0 = blockIdx.y * 128, col0 = blockIdx.x * 128;
    const int g = lane >> 4, l16 = lane & 15;

    // LDS byte offsets (flat LDS aperture: low 32 bits = workgroup offset)
    const unsigned ldsA0 = (unsigned)(size_t)(void*)&As[0][0];
    const unsigned ldsA1 = (unsigned)(size_t)(void*)&As[1][0];
    const unsigned ldsB0 = (unsigned)(size_t)(void*)&Bs[0][0];
    const unsigned ldsB1 = (unsigned)(size_t)(void*)&Bs[1][0];
    const unsigned long long gaBase =
        (unsigned long long)(size_t)(A + (size_t)row0 * Kdim);
    const unsigned long long gbBase =
        (unsigned long long)(size_t)(W + (size_t)col0 * Kdim);

    // D# group1: workgroup_mask=0, data_size=2B, pad_enable=1,
    // pad_interval=3 (16 DWORDs), pad_amount=3 (4 DWORDs),
    // tensor dims == tile dims (32 x 128) so OOB never clips,
    // tensor_dim0_stride = Kdim elements.
    uint8v g1;
    g1[0] = (1u << 16) | (1u << 20) | (3u << 22) | (3u << 25);
    g1[1] = (32u & 0xFFFFu) << 16;   // tensor_dim0[15:0]
    g1[2] = (128u << 16);            // tensor_dim0[31:16]=0 | tensor_dim1[15:0]
    g1[3] = (32u << 16);             // tensor_dim1[31:16]=0 | tile_dim0=32
    g1[4] = 128u;                    // tile_dim1=128 | tile_dim2=0
    g1[5] = (unsigned)Kdim;          // tensor_dim0_stride[31:0]
    g1[6] = 0u;                      // stride hi | tensor_dim1_stride lo
    g1[7] = 0u;

    auto tdm_issue = [&](int ktile, int buf) {
        unsigned long long ga = gaBase + (unsigned long long)ktile * 64u;
        unsigned long long gb = gbBase + (unsigned long long)ktile * 64u;
        uint4v d0a, d0b;
        d0a[0] = 1u;                                 // count=1 (valid), user mode
        d0a[1] = buf ? ldsA1 : ldsA0;
        d0a[2] = (unsigned)ga;
        d0a[3] = ((unsigned)(ga >> 32) & 0x01FFFFFFu) | 0x80000000u; // type=2
        d0b[0] = 1u;
        d0b[1] = buf ? ldsB1 : ldsB0;
        d0b[2] = (unsigned)gb;
        d0b[3] = ((unsigned)(gb >> 32) & 0x01FFFFFFu) | 0x80000000u;
        asm volatile("tensor_load_to_lds %0, %1" :: "s"(d0a), "s"(g1) : "memory");
        asm volatile("tensor_load_to_lds %0, %1" :: "s"(d0b), "s"(g1) : "memory");
    };

    v8f c[2][4];
#pragma unroll
    for (int i = 0; i < 2; i++)
#pragma unroll
        for (int j = 0; j < 4; j++) c[i][j] = {};

    const int nk = Kdim >> 5;
    if (wid == 0) tdm_issue(0, 0);                 // prologue: tile 0 -> buf 0
    for (int i = 0; i < nk; i++) {
        const int cur = i & 1;
        if (wid == 0) __builtin_amdgcn_s_wait_tensorcnt(0);  // tile i landed
        __syncthreads();                           // tile i visible; buf^1 dead
        if (wid == 0 && (i + 1) < nk) tdm_issue(i + 1, cur ^ 1); // overlap DMA

        const bf16* Ab = &As[cur][0];
        const bf16* Bb = &Bs[cur][0];
        v16bf a[2], b[4];
#pragma unroll
        for (int ii = 0; ii < 2; ii++) {           // A-frag: one 16-half run
            int m = wm * 32 + ii * 16 + l16;
            ((bf16x8*)&a[ii])[0] = *(const bf16x8*)&Ab[m * 40 + g * 16];
            ((bf16x8*)&a[ii])[1] = *(const bf16x8*)&Ab[m * 40 + g * 16 + 8];
        }
#pragma unroll
        for (int j = 0; j < 4; j++) {              // B-frag: one 16-half run
            int n = wn * 64 + j * 16 + l16;
            ((bf16x8*)&b[j])[0] = *(const bf16x8*)&Bb[n * 40 + g * 16];
            ((bf16x8*)&b[j])[1] = *(const bf16x8*)&Bb[n * 40 + g * 16 + 8];
        }
#pragma unroll
        for (int ii = 0; ii < 2; ii++)
#pragma unroll
            for (int j = 0; j < 4; j++)
                c[ii][j] = __builtin_amdgcn_wmma_f32_16x16x32_bf16(
                    false, a[ii], false, b[j], (short)0, c[ii][j], false, false);
    }

    // epilogue: element (m,n) -> lane = n + 16*(m/8), vgpr = m%8
#pragma unroll
    for (int i = 0; i < 2; i++) {
#pragma unroll
        for (int j = 0; j < 4; j++) {
            int colg = col0 + wn * 64 + j * 16 + l16;
#pragma unroll
            for (int vv = 0; vv < 8; vv++) {
                int rowg = row0 + wm * 32 + i * 16 + vv + 8 * g;
                float acc = c[i][j][vv];
                size_t o = (size_t)rowg * Ndim + colg;
                if (mode == 0)      outb[o] = (bf16)acc;
                else if (mode == 1) resid[o] = resid[o] + acc + bias[colg];
                else if (mode == 2) outb[o] = (bf16)gelu_f(acc + bias[colg]);
                else {              // mode 3: V in [B, D, T] (feature-major)
                    int bb = rowg / T_, tt = rowg % T_;
                    outb[((size_t)bb * D_ + colg) * T_ + tt] = (bf16)acc;
                }
            }
        }
    }
}

// ------------------------- Flash attention ---------------------------------
// One wave handles 16 queries of one (b,h). Streams keys in 32-wide tiles:
// scores via 2x (chained 16x16x32 bf16 WMMA), online softmax, P->LDS bounce
// (D-layout -> A-layout), P@V with V pre-transposed to [B,D,T] so every
// fragment load is a contiguous 16-half run.
__global__ __launch_bounds__(256)
void attn_k(const bf16* __restrict__ q, const bf16* __restrict__ k,
            const bf16* __restrict__ vt, bf16* __restrict__ ctx) {
    __shared__ __align__(16) bf16 P[8][16 * 40];   // per-wave P tile (16x32, pad 40)
    const int tid = threadIdx.x, lane = tid & 31, wid = tid >> 5;
    const int g = lane >> 4, l16 = lane & 15;
    const int idx = blockIdx.x * 8 + wid;          // 4096 wave tasks total
    const int qt = idx & 127;                      // T/16 = 128 query tiles
    const int bh = idx >> 7;
    const int hh = bh & (H_ - 1);
    const int b = bh >> 4;

    // Q fragments (A-layout), rows qt*16 + l16, hd in 2 slabs of 32
    v16bf qf[2];
    {
        int qrow = qt * 16 + l16;
#pragma unroll
        for (int g2 = 0; g2 < 2; g2++) {
            const bf16* p = q + (size_t)(b * T_ + qrow) * D_ + hh * HD_ + g2 * 32 + g * 16;
            ((bf16x8*)&qf[g2])[0] = *(const bf16x8*)p;
            ((bf16x8*)&qf[g2])[1] = *(const bf16x8*)(p + 8);
        }
    }
    // V row pointers in [B,D,T]: lane = hd column j*16+l16 of this head
    const bf16* vrow[4];
#pragma unroll
    for (int j = 0; j < 4; j++)
        vrow[j] = vt + ((size_t)b * D_ + hh * HD_ + j * 16 + l16) * T_;

    v8f o[4];
#pragma unroll
    for (int j = 0; j < 4; j++) o[j] = {};
    float mrow[8], lrow[8];
#pragma unroll
    for (int vv = 0; vv < 8; vv++) { mrow[vv] = -3.0e38f; lrow[vv] = 0.f; }

    const int lastq = qt * 16 + 15;
    const int nkt = lastq / 32 + 1;
    bf16* pl = &P[wid][0];

    for (int kt = 0; kt < nkt; kt++) {
        const int kbase = kt * 32;
        v8f s[2];
#pragma unroll
        for (int ss = 0; ss < 2; ss++) {
            // K fragments (B-layout): lane = key column, 16 contiguous hd halves
            int key = kbase + ss * 16 + l16;
            const bf16* p0 = k + (size_t)(b * T_ + key) * D_ + hh * HD_ + g * 16;
            v16bf kf0, kf1;
            ((bf16x8*)&kf0)[0] = *(const bf16x8*)p0;
            ((bf16x8*)&kf0)[1] = *(const bf16x8*)(p0 + 8);
            ((bf16x8*)&kf1)[0] = *(const bf16x8*)(p0 + 32);
            ((bf16x8*)&kf1)[1] = *(const bf16x8*)(p0 + 40);
            v8f acc = {};
            acc = __builtin_amdgcn_wmma_f32_16x16x32_bf16(false, qf[0], false, kf0,
                                                          (short)0, acc, false, false);
            acc = __builtin_amdgcn_wmma_f32_16x16x32_bf16(false, qf[1], false, kf1,
                                                          (short)0, acc, false, false);
            s[ss] = acc;
        }
        // causal mask (then scale, like the reference), D-layout indexing
#pragma unroll
        for (int ss = 0; ss < 2; ss++)
#pragma unroll
            for (int vv = 0; vv < 8; vv++) {
                int qrow = qt * 16 + vv + 8 * g;
                int key = kbase + ss * 16 + l16;
                float val = s[ss][vv];
                s[ss][vv] = (key > qrow) ? -3.0e38f : val * 0.125f;  // 1/sqrt(64)
            }
        // online softmax across the 16 key-lanes of each half-wave group
#pragma unroll
        for (int vv = 0; vv < 8; vv++) {
            float mloc = fmaxf(s[0][vv], s[1][vv]);
            for (int d2 = 1; d2 < 16; d2 <<= 1) mloc = fmaxf(mloc, __shfl_xor(mloc, d2, 16));
            float mnew = fmaxf(mrow[vv], mloc);
            float p0 = __expf(s[0][vv] - mnew);
            float p1 = __expf(s[1][vv] - mnew);
            float rsum = p0 + p1;
            for (int d2 = 1; d2 < 16; d2 <<= 1) rsum += __shfl_xor(rsum, d2, 16);
            float alpha = __expf(mrow[vv] - mnew);
            lrow[vv] = lrow[vv] * alpha + rsum;
            mrow[vv] = mnew;
            s[0][vv] = p0; s[1][vv] = p1;
#pragma unroll
            for (int j = 0; j < 4; j++) o[j][vv] *= alpha;
        }
        // P (D-layout) -> per-wave LDS -> A-layout fragment. LDS is in-order
        // within a wave; wait DScnt before reading the freshly stored tile.
#pragma unroll
        for (int ss = 0; ss < 2; ss++)
#pragma unroll
            for (int vv = 0; vv < 8; vv++)
                pl[(vv + 8 * g) * 40 + ss * 16 + l16] = (bf16)s[ss][vv];
        asm volatile("s_wait_dscnt 0" ::: "memory");
        v16bf pf;
        ((bf16x8*)&pf)[0] = *(const bf16x8*)&pl[l16 * 40 + g * 16];
        ((bf16x8*)&pf)[1] = *(const bf16x8*)&pl[l16 * 40 + g * 16 + 8];

        // V fragments (B-layout): 16 contiguous key-halves per lane from [B,D,T]
#pragma unroll
        for (int j = 0; j < 4; j++) {
            const bf16* vp = vrow[j] + kbase + g * 16;
            v16bf vf;
            ((bf16x8*)&vf)[0] = *(const bf16x8*)vp;
            ((bf16x8*)&vf)[1] = *(const bf16x8*)(vp + 8);
            o[j] = __builtin_amdgcn_wmma_f32_16x16x32_bf16(false, pf, false, vf,
                                                           (short)0, o[j], false, false);
        }
    }

    // normalize + store ctx as [B,T,H,HD] bf16
#pragma unroll
    for (int vv = 0; vv < 8; vv++) {
        float inv = 1.0f / lrow[vv];
        int qrow = qt * 16 + vv + 8 * g;
#pragma unroll
        for (int j = 0; j < 4; j++)
            ctx[(size_t)(b * T_ + qrow) * D_ + hh * HD_ + j * 16 + l16] =
                (bf16)(o[j][vv] * inv);
    }
}

// ------------------------------- logits ------------------------------------
// One wave per vocab row; both batch rows accumulated together; W_emb stays f32
// (single streaming pass, pure bandwidth).
__global__ __launch_bounds__(256)
void logits_k(const float* __restrict__ hf, const float* __restrict__ We,
              float* __restrict__ out) {
    int lane = threadIdx.x & 31, wid = threadIdx.x >> 5;
    int row = blockIdx.x * 8 + wid;
    if (row >= V_) return;
    const float* wr = We + (size_t)row * D_;
    float a0 = 0.f, a1 = 0.f;
#pragma unroll 4
    for (int i = 0; i < D_ / 32; i++) {
        int d = lane + i * 32;
        float w = wr[d];
        a0 += hf[d] * w;
        a1 += hf[D_ + d] * w;
    }
    for (int m = 1; m < 32; m <<= 1) { a0 += __shfl_xor(a0, m, 32); a1 += __shfl_xor(a1, m, 32); }
    if (lane == 0) { out[row] = a0; out[V_ + row] = a1; }
}

// ------------------------------- driver ------------------------------------
extern "C" void kernel_launch(void* const* d_in, const int* in_sizes, int n_in,
                              void* d_out, int out_size, void* d_ws, size_t ws_size,
                              hipStream_t stream) {
    (void)in_sizes; (void)n_in; (void)out_size; (void)ws_size;
    const int*   x   = (const int*)d_in[0];
    const float* We  = (const float*)d_in[1];
    const float* pe  = (const float*)d_in[2];
    const float* Wq  = (const float*)d_in[3];
    const float* Wk  = (const float*)d_in[4];
    const float* Wv  = (const float*)d_in[5];
    const float* Wo  = (const float*)d_in[6];
    const float* bo  = (const float*)d_in[7];
    const float* n1s = (const float*)d_in[8];
    const float* n1b = (const float*)d_in[9];
    const float* n2s = (const float*)d_in[10];
    const float* n2b = (const float*)d_in[11];
    const float* W1  = (const float*)d_in[12];
    const float* b1  = (const float*)d_in[13];
    const float* W2  = (const float*)d_in[14];
    const float* b2  = (const float*)d_in[15];
    const float* fs  = (const float*)d_in[16];
    const float* fb  = (const float*)d_in[17];
    float* out = (float*)d_out;

    char* wsp = (char*)d_ws;
    auto alloc = [&](size_t bytes) -> char* {
        char* p = wsp;
        wsp += (bytes + 255) & ~(size_t)255;
        return p;
    };
    bf16*  Wqb = (bf16*)alloc((size_t)L_ * D_ * D_ * 2);
    bf16*  Wkb = (bf16*)alloc((size_t)L_ * D_ * D_ * 2);
    bf16*  Wvb = (bf16*)alloc((size_t)L_ * D_ * D_ * 2);
    bf16*  Wob = (bf16*)alloc((size_t)L_ * D_ * D_ * 2);
    bf16*  W1b = (bf16*)alloc((size_t)L_ * DFF_ * D_ * 2);
    bf16*  W2b = (bf16*)alloc((size_t)L_ * D_ * DFF_ * 2);
    float* h   = (float*)alloc((size_t)M_ * D_ * 4);
    bf16*  xn  = (bf16*)alloc((size_t)M_ * D_ * 2);
    bf16*  qb  = (bf16*)alloc((size_t)M_ * D_ * 2);
    bf16*  kb  = (bf16*)alloc((size_t)M_ * D_ * 2);
    bf16*  vb  = (bf16*)alloc((size_t)M_ * D_ * 2);   // [B, D, T] layout
    bf16*  cb  = (bf16*)alloc((size_t)M_ * D_ * 2);
    bf16*  ffb = (bf16*)alloc((size_t)M_ * DFF_ * 2);
    float* hf  = (float*)alloc((size_t)B_ * D_ * 4);

    // weights -> bf16 (192 MB total; thereafter the per-layer weight stream
    // is L2-resident on the 192 MB global L2)
    const long long nw = (long long)L_ * D_ * D_;
    const long long nf = (long long)L_ * DFF_ * D_;
    cvt_f32_bf16_k<<<4096, 256, 0, stream>>>(Wq, Wqb, nw);
    cvt_f32_bf16_k<<<4096, 256, 0, stream>>>(Wk, Wkb, nw);
    cvt_f32_bf16_k<<<4096, 256, 0, stream>>>(Wv, Wvb, nw);
    cvt_f32_bf16_k<<<4096, 256, 0, stream>>>(Wo, Wob, nw);
    cvt_f32_bf16_k<<<4096, 256, 0, stream>>>(W1, W1b, nf);
    cvt_f32_bf16_k<<<4096, 256, 0, stream>>>(W2, W2b, nf);

    embed_k<<<4096, 256, 0, stream>>>(x, We, pe, h);

    dim3 blk(256);
    dim3 gD(D_ / 128, M_ / 128);      // N=1024 GEMMs
    dim3 gF(DFF_ / 128, M_ / 128);    // N=4096 GEMM
    for (int l = 0; l < L_; l++) {
        ln_bf16_k<<<M_, blk, 0, stream>>>(h, n1s + (size_t)l * D_, n1b + (size_t)l * D_, xn);
        gemm_wmma_k<<<gD, blk, 0, stream>>>(xn, Wqb + (size_t)l * D_ * D_,
                                            nullptr, nullptr, qb, M_, D_, D_, 0);
        gemm_wmma_k<<<gD, blk, 0, stream>>>(xn, Wkb + (size_t)l * D_ * D_,
                                            nullptr, nullptr, kb, M_, D_, D_, 0);
        gemm_wmma_k<<<gD, blk, 0, stream>>>(xn, Wvb + (size_t)l * D_ * D_,
                                            nullptr, nullptr, vb, M_, D_, D_, 3);
        attn_k<<<(B_ * H_ * (T_ / 16)) / 8, blk, 0, stream>>>(qb, kb, vb, cb);
        gemm_wmma_k<<<gD, blk, 0, stream>>>(cb, Wob + (size_t)l * D_ * D_,
                                            bo + (size_t)l * D_, h, nullptr, M_, D_, D_, 1);
        ln_bf16_k<<<M_, blk, 0, stream>>>(h, n2s + (size_t)l * D_, n2b + (size_t)l * D_, xn);
        gemm_wmma_k<<<gF, blk, 0, stream>>>(xn, W1b + (size_t)l * DFF_ * D_,
                                            b1 + (size_t)l * DFF_, nullptr, ffb,
                                            M_, DFF_, D_, 2);
        gemm_wmma_k<<<gD, blk, 0, stream>>>(ffb, W2b + (size_t)l * D_ * DFF_,
                                            b2 + (size_t)l * D_, h, nullptr,
                                            M_, D_, DFF_, 1);
    }
    ln_final_k<<<B_, blk, 0, stream>>>(h, fs, fb, hf);
    logits_k<<<(V_ + 7) / 8, blk, 0, stream>>>(hf, We, out);
}
